// Zencoder_24988119728709
// MI455X (gfx1250) — matrix operand order repaired and checked by reading, not compiled, not run on hardware
//
#include <hip/hip_runtime.h>
#include <stdint.h>

typedef __attribute__((ext_vector_type(16))) __bf16 v16bf;
typedef __attribute__((ext_vector_type(8)))  float  v8f;

// ---------------------------------------------------------------- helpers

__device__ __forceinline__ uint16_t f2bf(float f) {
    uint32_t u = __float_as_uint(f);
    uint32_t r = u + 0x7FFFu + ((u >> 16) & 1u);   // round-to-nearest-even
    return (uint16_t)(r >> 16);
}

__device__ __forceinline__ v16bf load16bf(const uint16_t* p) {
    union { uint4 u[2]; v16bf v; } t;
    const uint4* q = (const uint4*)p;
    t.u[0] = q[0];
    t.u[1] = q[1];
    return t.v;
}

__device__ __forceinline__ v16bf zero16bf() {
    union { uint4 u[2]; v16bf v; } t;
    t.u[0] = make_uint4(0u, 0u, 0u, 0u);
    t.u[1] = make_uint4(0u, 0u, 0u, 0u);
    return t.v;
}

__device__ __forceinline__ v8f wmma_bf16(v16bf a, v16bf b, v8f c) {
    // D(f32 16x16) = A(bf16 16x32) * B(bf16 32x16) + C
    return __builtin_amdgcn_wmma_f32_16x16x32_bf16(false, a, false, b,
                                                   (short)0, c, false, false);
}

__device__ __forceinline__ int refl(int i, int n) {
    i = (i < 0) ? -i : i;
    return (i >= n) ? (2 * n - 2 - i) : i;
}

// ---------------------------------------------------------------- weight packing
// out layout: [O/16][Ktiles][lane:32][j:16] bf16, element (lane,j) holds A(m,k) with
//   m = mtile*16 + (lane&15),  k = kt*32 + 16*(lane>>4) + j
// K ordering: k = tap*I + c. mode 0: tap=ky*3+kx (3x3 conv). mode 1: convT parity class.
__global__ void pack_w_kernel(const float* __restrict__ w, uint16_t* __restrict__ out,
                              int O, int I, int K, int mode, int cls, int total) {
    int idx = blockIdx.x * 256 + threadIdx.x;
    if (idx >= total) return;
    int j     = idx & 15;
    int lane  = (idx >> 4) & 31;
    int rest  = idx >> 9;
    int Kt    = (K + 31) >> 5;
    int kt    = rest % Kt;
    int mtile = rest / Kt;
    int m     = mtile * 16 + (lane & 15);
    int halfA = lane >> 4;
    int k     = kt * 32 + 16 * halfA + j;
    float val = 0.f;
    if (k < K && m < O) {
        int t = k / I, c = k - t * I;
        int ky, kx;
        if (mode == 0) { ky = t / 3; kx = t - ky * 3; }
        else {
            int py = cls >> 1, px = cls & 1;
            int ntx = px ? 2 : 1;
            int tx = t % ntx, ty = t / ntx;
            ky = py ? (ty ? 2 : 0) : 1;
            kx = px ? (tx ? 2 : 0) : 1;
        }
        val = w[(((size_t)m * I + c) * 3 + ky) * 3 + kx];
    }
    out[idx] = f2bf(val);
}

// ---------------------------------------------------------------- generic implicit-GEMM conv (compile-time geometry)
// x: bf16 NHWC [8,Hin,Win,Cin]; y: f32 NCHW [8,Cout,Hout,Wout]
// Each wave: 16 out-channels x 64 pixels (4 accumulators reuse one A load).
template<int Cin, int Hin, int Win, int Cout, int Hout, int Wout,
         int Ksz, int STRIDE, int PAD, int REFLECT>
__global__ void conv_wmma_t(const uint16_t* __restrict__ x,
                            const uint16_t* __restrict__ wpk,
                            const float* __restrict__ bias,
                            float* __restrict__ y) {
    constexpr int Kt = (Ksz + 31) / 32;
    constexpr int NT = 4;
    int wave = threadIdx.x >> 5;
    int lane = threadIdx.x & 31;
    int half = lane >> 4;
    int nIdx = lane & 15;
    int b     = blockIdx.z;
    int mtile = blockIdx.y;
    int pbase = (blockIdx.x * 8 + wave) * (16 * NT) + nIdx;

    int oy[NT], ox[NT];
#pragma unroll
    for (int t = 0; t < NT; ++t) {
        int p = pbase + 16 * t;
        oy[t] = p / Wout;          // Wout: power of two -> shift
        ox[t] = p % Wout;
    }

    v8f acc[NT] = {};
    const uint16_t* wp = wpk + ((size_t)mtile * Kt) * 512 + lane * 16;

    for (int kt = 0; kt < Kt; ++kt, wp += 512) {
        __builtin_prefetch(wp + 512, 0, 0);
        v16bf a = load16bf(wp);
        int kbase = kt * 32;
        int tap = kbase / Cin;                       // Cin: power of two -> shift
        int c0  = (kbase & (Cin - 1)) + (half << 4);
        int ky = tap / 3, kx = tap - ky * 3;
#pragma unroll
        for (int t = 0; t < NT; ++t) {
            int iy = oy[t] * STRIDE - PAD + ky;
            int ix = ox[t] * STRIDE - PAD + kx;
            if (REFLECT) { iy = refl(iy, Hin); ix = refl(ix, Win); }
            v16bf bm;
            if (iy >= 0 && iy < Hin && ix >= 0 && ix < Win)
                bm = load16bf(x + (((size_t)b * Hin + iy) * Win + ix) * Cin + c0);
            else
                bm = zero16bf();
            acc[t] = wmma_bf16(a, bm, acc[t]);
        }
    }

    // D layout: VGPR r -> M = r + 8*half, N = lane&15
#pragma unroll
    for (int t = 0; t < NT; ++t)
        for (int r = 0; r < 8; ++r) {
            int m = mtile * 16 + r + (half << 3);
            y[(((size_t)b * Cout + m) * Hout + oy[t]) * Wout + ox[t]] = acc[t][r] + bias[m];
        }
}

// ---------------------------------------------------------------- conv0: Cin=3, K=27, reflect pad (scalar-gather B)
__global__ void conv0_kernel(const uint16_t* __restrict__ x,
                             const uint16_t* __restrict__ wpk,
                             const float* __restrict__ bias,
                             float* __restrict__ y) {
    int wave = threadIdx.x >> 5;
    int lane = threadIdx.x & 31;
    int half = lane >> 4;
    int nIdx = lane & 15;
    int b     = blockIdx.z;
    int mtile = blockIdx.y;
    int p = (blockIdx.x * 8 + wave) * 16 + nIdx;
    int oy = p >> 8, ox = p & 255;

    v8f acc = {};
    const uint16_t* wp = wpk + (size_t)mtile * 512 + lane * 16;
    v16bf a = load16bf(wp);

    union { uint16_t s[16]; v16bf v; } t;
#pragma unroll
    for (int j = 0; j < 16; ++j) {
        int k = (half << 4) + j;
        uint16_t vv = 0;
        if (k < 27) {
            int tap = k / 3, c = k - tap * 3;
            int ky = tap / 3, kx = tap - ky * 3;
            int iy = refl(oy - 1 + ky, 256);
            int ix = refl(ox - 1 + kx, 256);
            vv = x[(((size_t)b * 256 + iy) * 256 + ix) * 3 + c];
        }
        t.s[j] = vv;
    }
    acc = wmma_bf16(a, t.v, acc);

    for (int r = 0; r < 8; ++r) {
        int m = mtile * 16 + r + (half << 3);
        y[(((size_t)b * 32 + m) * 256 + oy) * 256 + ox] = acc[r] + bias[m];
    }
}

// ---------------------------------------------------------------- transposed conv (stride 2): one parity class per instantiation
// x: bf16 NHWC [8,64,64,128]; y: f32 NCHW [8,256,128,128]
template<int CLS>
__global__ void convT_t(const uint16_t* __restrict__ x,
                        const uint16_t* __restrict__ wpk,   // class base
                        const float* __restrict__ bias,
                        float* __restrict__ y) {
    constexpr int PY = CLS >> 1, PX = CLS & 1;
    constexpr int NTX = PX ? 2 : 1, NTY = PY ? 2 : 1;
    constexpr int Kt = (NTY * NTX * 128) / 32;
    constexpr int NT = 4;
    int wave = threadIdx.x >> 5;
    int lane = threadIdx.x & 31;
    int half = lane >> 4;
    int nIdx = lane & 15;
    int b     = blockIdx.z;
    int mtile = blockIdx.y;
    int pbase = (blockIdx.x * 8 + wave) * (16 * NT) + nIdx;

    int ojy[NT], ojx[NT];
#pragma unroll
    for (int t = 0; t < NT; ++t) {
        int p = pbase + 16 * t;
        ojy[t] = p >> 6;
        ojx[t] = p & 63;
    }

    v8f acc[NT] = {};
    const uint16_t* wp = wpk + ((size_t)mtile * Kt) * 512 + lane * 16;
    for (int kt = 0; kt < Kt; ++kt, wp += 512) {
        v16bf a = load16bf(wp);
        int tt = kt >> 2;                        // kbase / 128
        int c0 = ((kt & 3) << 5) + (half << 4);
        int tx = (NTX == 2) ? (tt & 1) : 0;
        int ty = (NTX == 2) ? (tt >> 1) : tt;
        int diy = PY ? ty : 0;
        int dix = PX ? tx : 0;
#pragma unroll
        for (int t = 0; t < NT; ++t) {
            int iy = ojy[t] + diy, ix = ojx[t] + dix;
            v16bf bm;
            if (iy < 64 && ix < 64)
                bm = load16bf(x + (((size_t)b * 64 + iy) * 64 + ix) * 128 + c0);
            else
                bm = zero16bf();
            acc[t] = wmma_bf16(a, bm, acc[t]);
        }
    }
#pragma unroll
    for (int t = 0; t < NT; ++t) {
        int oy = 2 * ojy[t] + PY, ox = 2 * ojx[t] + PX;
        for (int r = 0; r < 8; ++r) {
            int m = mtile * 16 + r + (half << 3);
            y[(((size_t)b * 256 + m) * 128 + oy) * 128 + ox] = acc[t][r] + bias[m];
        }
    }
}

// ---------------------------------------------------------------- conv3 (reflect) fused with tanh + segment pooling
// x: bf16 NHWC [8,128,128,256]; sums: f32 [8][19][512]
// Weight slab for this mtile (72KB) staged in LDS, shared by all 8 waves.
__global__ void conv3_pool_kernel(const uint16_t* __restrict__ x,
                                  const uint16_t* __restrict__ wpk,
                                  const float* __restrict__ bias,
                                  const int* __restrict__ lab,
                                  float* __restrict__ sums) {
    constexpr int Kt = 72;                 // K = 2304
    constexpr int NT = 4;
    __shared__ uint16_t wlds[Kt * 512];    // 73728 B
    __shared__ float lsum[19 * 16];

    int b     = blockIdx.z;
    int mtile = blockIdx.y;

    // cooperative copy of the packed-weight slab into LDS
    {
        const uint4* src = (const uint4*)(wpk + (size_t)mtile * (Kt * 512));
        uint4* dst = (uint4*)wlds;
#pragma unroll
        for (int i = threadIdx.x; i < Kt * 512 / 8; i += 256) dst[i] = src[i];
    }
    for (int i = threadIdx.x; i < 304; i += 256) lsum[i] = 0.f;
    __syncthreads();

    int wave = threadIdx.x >> 5;
    int lane = threadIdx.x & 31;
    int half = lane >> 4;
    int nIdx = lane & 15;
    int pbase = (blockIdx.x * 8 + wave) * (16 * NT) + nIdx;

    int oy[NT], ox[NT];
#pragma unroll
    for (int t = 0; t < NT; ++t) {
        int p = pbase + 16 * t;
        oy[t] = p >> 7;
        ox[t] = p & 127;
    }

    v8f acc[NT] = {};
    const uint16_t* wp = wlds + lane * 16;
    for (int kt = 0; kt < Kt; ++kt, wp += 512) {
        v16bf a = load16bf(wp);
        int kbase = kt << 5;
        int tap = kbase >> 8;                      // / 256
        int c0  = (kbase & 255) + (half << 4);
        int ky = tap / 3, kx = tap - ky * 3;
#pragma unroll
        for (int t = 0; t < NT; ++t) {
            int iy = refl(oy[t] - 1 + ky, 128);
            int ix = refl(ox[t] - 1 + kx, 128);
            v16bf bm = load16bf(x + (((size_t)b * 128 + iy) * 128 + ix) * 256 + c0);
            acc[t] = wmma_bf16(a, bm, acc[t]);
        }
    }

#pragma unroll
    for (int t = 0; t < NT; ++t) {
        int s = lab[((size_t)b << 14) + pbase + 16 * t];
        for (int r = 0; r < 8; ++r) {
            int mi = r + (half << 3);
            float v = tanhf(acc[t][r] + bias[mtile * 16 + mi]);
            atomicAdd(&lsum[s * 16 + mi], v);
        }
    }
    __syncthreads();
    for (int i = threadIdx.x; i < 304; i += 256)
        atomicAdd(&sums[((size_t)b * 19 + (i >> 4)) * 512 + mtile * 16 + (i & 15)], lsum[i]);
}

// ---------------------------------------------------------------- instance-norm stats
__global__ void stats_kernel(const float* __restrict__ y, float* __restrict__ mean,
                             float* __restrict__ rstd, int HW) {
    __shared__ float s1[256], s2[256];
    const float* p = y + (size_t)blockIdx.x * HW;
    float a = 0.f, b = 0.f;
    for (int i = threadIdx.x; i < HW; i += 256) { float v = p[i]; a += v; b += v * v; }
    s1[threadIdx.x] = a; s2[threadIdx.x] = b;
    __syncthreads();
    for (int off = 128; off; off >>= 1) {
        if (threadIdx.x < off) {
            s1[threadIdx.x] += s1[threadIdx.x + off];
            s2[threadIdx.x] += s2[threadIdx.x + off];
        }
        __syncthreads();
    }
    if (threadIdx.x == 0) {
        float m = s1[0] / (float)HW;
        float v = s2[0] / (float)HW - m * m;
        mean[blockIdx.x] = m;
        rstd[blockIdx.x] = rsqrtf(fmaxf(v, 0.f) + 1e-5f);
    }
}

// normalize + LeakyReLU + f32 NCHW -> bf16 NHWC (feeds next conv)
template<int C, int H, int W>
__global__ void apply_t(const float* __restrict__ y, const float* __restrict__ mean,
                        const float* __restrict__ rstd, uint16_t* __restrict__ out) {
    constexpr int HW = H * W;
    int idx = blockIdx.x * 256 + threadIdx.x;
    if (idx >= 8 * C * HW) return;
    int bc = idx / HW;             // power-of-two -> shift
    int hw = idx & (HW - 1);
    int b = bc / C, c = bc & (C - 1);
    float v = (y[idx] - mean[bc]) * rstd[bc];
    v = (v >= 0.f) ? v : 0.2f * v;
    out[(((size_t)b * H + (hw / W)) * W + (hw & (W - 1))) * C + c] = f2bf(v);
}

// input f32 NCHW [8,3,256,256] -> bf16 NHWC
__global__ void cvt_in_kernel(const float* __restrict__ in, uint16_t* __restrict__ out) {
    int idx = blockIdx.x * 256 + threadIdx.x;
    if (idx >= 8 * 3 * 65536) return;
    int b = idx / (3 * 65536);
    int r = idx - b * (3 * 65536);
    int c = r >> 16;
    int hw = r & 65535;
    out[((size_t)b * 65536 + hw) * 3 + c] = f2bf(in[idx]);
}

// segmap one-hot -> per-(downsampled)pixel label + per-segment counts
__global__ void labels_kernel(const float* __restrict__ seg, int* __restrict__ lab,
                              float* __restrict__ cnt) {
    int idx = blockIdx.x * 256 + threadIdx.x;
    if (idx >= 8 * 128 * 128) return;
    int b = idx >> 14;
    int hw = idx & 16383;
    int h = hw >> 7, w = hw & 127;
    int s = 0;
    for (int t = 0; t < 19; ++t) {
        if (seg[((size_t)(b * 19 + t) << 16) + (h << 9) + (w << 1)] != 0.f) { s = t; break; }
    }
    lab[idx] = s;
    atomicAdd(&cnt[b * 19 + s], 1.0f);
}

__global__ void zero_kernel(float* __restrict__ p, int n) {
    int i = blockIdx.x * 256 + threadIdx.x;
    if (i < n) p[i] = 0.f;
}

__global__ void finalize_kernel(const float* __restrict__ sums, const float* __restrict__ cnt,
                                float* __restrict__ out) {
    int idx = blockIdx.x * 256 + threadIdx.x;
    if (idx >= 8 * 19 * 512) return;
    float c = cnt[idx >> 9];
    out[idx] = (c > 0.f) ? sums[idx] / fmaxf(c, 1.f) : 0.f;
}

// ---------------------------------------------------------------- launch

static inline int cdiv(int a, int b) { return (a + b - 1) / b; }

extern "C" void kernel_launch(void* const* d_in, const int* in_sizes, int n_in,
                              void* d_out, int out_size, void* d_ws, size_t ws_size,
                              hipStream_t stream) {
    (void)in_sizes; (void)n_in; (void)out_size; (void)ws_size;
    const float* inp = (const float*)d_in[0];   // [8,3,256,256]
    const float* seg = (const float*)d_in[1];   // [8,19,256,256]
    const float* w0  = (const float*)d_in[2];   const float* b0 = (const float*)d_in[3];
    const float* w1  = (const float*)d_in[4];   const float* b1 = (const float*)d_in[5];
    const float* w2  = (const float*)d_in[6];   const float* b2 = (const float*)d_in[7];
    const float* wt  = (const float*)d_in[8];   const float* bt = (const float*)d_in[9];
    const float* w3  = (const float*)d_in[10];  const float* b3 = (const float*)d_in[11];
    float* out = (float*)d_out;

    // ---- workspace carve (aliased across stages), all offsets 256B aligned
    char* ws = (char*)d_ws;
    size_t off = 0;
    auto carve = [&](size_t bytes) { char* p = ws + off; off += (bytes + 255) & ~(size_t)255; return p; };
    float*    ybuf = (float*)   carve(134217728);   // f32 conv-out, reused every stage (max = convT out)
    uint16_t* regA = (uint16_t*)carve(67108864);    // bf16 act: a_in -> a1 -> at
    uint16_t* regB = (uint16_t*)carve(33554432);    // bf16 act: a0 -> a2
    uint16_t* w0p  = (uint16_t*)carve(2048);
    uint16_t* w1p  = (uint16_t*)carve(36864);
    uint16_t* w2p  = (uint16_t*)carve(147456);
    uint16_t* wtp  = (uint16_t*)carve(589824);      // 4 parity classes, contiguous
    uint16_t* w3p  = (uint16_t*)carve(2359296);
    float*    mean = (float*)   carve(4096 * 4);
    float*    rstd = (float*)   carve(4096 * 4);
    int*      lab  = (int*)     carve(8 * 128 * 128 * 4);
    float*    sums = (float*)   carve(77824 * 4);
    float*    cnt  = (float*)   carve(8 * 19 * 4);  // adjacent to sums

    // ---- pack weights into WMMA A-register layout
    {
        int t0 = 2 * 1 * 512;
        pack_w_kernel<<<cdiv(t0, 256), 256, 0, stream>>>(w0, w0p, 32, 3, 27, 0, 0, t0);
        int t1 = 4 * 9 * 512;
        pack_w_kernel<<<cdiv(t1, 256), 256, 0, stream>>>(w1, w1p, 64, 32, 288, 0, 0, t1);
        int t2 = 8 * 18 * 512;
        pack_w_kernel<<<cdiv(t2, 256), 256, 0, stream>>>(w2, w2p, 128, 64, 576, 0, 0, t2);
        int kcls[4] = {128, 256, 256, 512};
        int eoff[4] = {0, 32768, 98304, 163840};
        for (int c = 0; c < 4; ++c) {
            int tc = 16 * (kcls[c] >> 5) * 512;
            pack_w_kernel<<<cdiv(tc, 256), 256, 0, stream>>>(wt, wtp + eoff[c], 256, 128, kcls[c], 1, c, tc);
        }
        int t3 = 32 * 72 * 512;
        pack_w_kernel<<<cdiv(t3, 256), 256, 0, stream>>>(w3, w3p, 512, 256, 2304, 0, 0, t3);
    }

    // ---- init sums+cnt (contiguous region), labels
    zero_kernel<<<cdiv(77824 + 8 * 19, 256), 256, 0, stream>>>(sums, 77824 + 8 * 19);
    labels_kernel<<<cdiv(8 * 128 * 128, 256), 256, 0, stream>>>(seg, lab, cnt);

    // ---- input -> bf16 NHWC
    cvt_in_kernel<<<cdiv(8 * 3 * 65536, 256), 256, 0, stream>>>(inp, regA);

    // ---- conv0 (reflect pad, Cin=3, scalar-gather path) -> ybuf [8,32,256,256]
    conv0_kernel<<<dim3(65536 / 128, 2, 8), 256, 0, stream>>>(regA, w0p, b0, ybuf);
    stats_kernel<<<8 * 32, 256, 0, stream>>>(ybuf, mean, rstd, 65536);
    apply_t<32, 256, 256><<<cdiv(8 * 32 * 65536, 256), 256, 0, stream>>>(ybuf, mean, rstd, regB);

    // ---- conv1 stride2 -> ybuf [8,64,128,128]
    conv_wmma_t<32, 256, 256, 64, 128, 128, 288, 2, 1, 0>
        <<<dim3(16384 / 512, 4, 8), 256, 0, stream>>>(regB, w1p, b1, ybuf);
    stats_kernel<<<8 * 64, 256, 0, stream>>>(ybuf, mean, rstd, 16384);
    apply_t<64, 128, 128><<<cdiv(8 * 64 * 16384, 256), 256, 0, stream>>>(ybuf, mean, rstd, regA);

    // ---- conv2 stride2 -> ybuf [8,128,64,64]
    conv_wmma_t<64, 128, 128, 128, 64, 64, 576, 2, 1, 0>
        <<<dim3(4096 / 512, 8, 8), 256, 0, stream>>>(regA, w2p, b2, ybuf);
    stats_kernel<<<8 * 128, 256, 0, stream>>>(ybuf, mean, rstd, 4096);
    apply_t<128, 64, 64><<<cdiv(8 * 128 * 4096, 256), 256, 0, stream>>>(ybuf, mean, rstd, regB);

    // ---- transposed conv (4 parity classes) -> ybuf [8,256,128,128]
    convT_t<0><<<dim3(4096 / 512, 16, 8), 256, 0, stream>>>(regB, wtp +      0, bt, ybuf);
    convT_t<1><<<dim3(4096 / 512, 16, 8), 256, 0, stream>>>(regB, wtp +  32768, bt, ybuf);
    convT_t<2><<<dim3(4096 / 512, 16, 8), 256, 0, stream>>>(regB, wtp +  98304, bt, ybuf);
    convT_t<3><<<dim3(4096 / 512, 16, 8), 256, 0, stream>>>(regB, wtp + 163840, bt, ybuf);
    stats_kernel<<<8 * 256, 256, 0, stream>>>(ybuf, mean, rstd, 16384);
    apply_t<256, 128, 128><<<cdiv(8 * 256 * 16384, 256), 256, 0, stream>>>(ybuf, mean, rstd, regA);

    // ---- conv3 (reflect, LDS-staged weights) + tanh + segment pooling (fused)
    conv3_pool_kernel<<<dim3(16384 / 512, 32, 8), 256, 0, stream>>>(regA, w3p, b3, lab, sums);

    // ---- out[b][s][c] = cnt>0 ? sums/max(cnt,1) : 0
    finalize_kernel<<<cdiv(77824, 256), 256, 0, stream>>>(sums, cnt, out);
}
